// STTransformerLayer_36309653520638
// MI455X (gfx1250) — matrix operand order repaired and checked by reading, not compiled, not run on hardware
//
#include <hip/hip_runtime.h>
#include <hip/hip_bf16.h>
#include <cstdint>

// ---------------------------------------------------------------------------
// ST-Transformer layer for MI455X (gfx1250, wave32, WMMA).
// Heavy NT GEMMs: 64x64 LDS-tiled, double-buffered, async global->LDS staging
//                 (global_load_async_to_lds + s_wait_asynccnt), 2 WMMA/wave/step.
// Attention batched GEMMs: direct-from-global WMMA kernel.
// ---------------------------------------------------------------------------

typedef _Float16 v16h __attribute__((ext_vector_type(16)));
typedef _Float16 h8   __attribute__((ext_vector_type(8)));
typedef _Float16 h4   __attribute__((ext_vector_type(4)));
typedef float    v8f  __attribute__((ext_vector_type(8)));
typedef int      ai2  __attribute__((vector_size(8)));    // async copy payload types
typedef int      ai4  __attribute__((vector_size(16)));

#define D_  512
#define H_  8
#define FF_ 2048
#define B_  8
#define T_  128
#define J_  32
#define M_  256
#define NTOK 32768            // B*T*J
#define TOKPERB 4096          // T*J

#if defined(__AMDGCN__) && __has_builtin(__builtin_amdgcn_global_load_async_to_lds_b128) && \
    __has_builtin(__builtin_amdgcn_s_wait_asynccnt)
#define USE_ASYNC 1
#define AS_GLOBAL __attribute__((address_space(1)))
#define AS_LOCAL  __attribute__((address_space(3)))
#else
#define USE_ASYNC 0
#endif

// ------------------------------ small kernels ------------------------------

__global__ void cvt_f32_f16(const float* __restrict__ in, _Float16* __restrict__ out, long long n) {
    long long i = (long long)blockIdx.x * blockDim.x + threadIdx.x;
    if (i < n) out[i] = (_Float16)in[i];
}

// FiLM: out[b, 0..1023] = silu(cond[b,:]) @ W.T + bias     (W: [1024, 512])
__global__ void film_kernel(const float* __restrict__ cond, const float* __restrict__ W,
                            const float* __restrict__ bias, float* __restrict__ out) {
    int o = blockIdx.x * 256 + threadIdx.x;   // 0..1023 (grid.x == 4)
    int b = blockIdx.y;                       // 0..7
    float acc = bias[o];
    const float* c = cond + b * D_;
    const float* w = W + (size_t)o * D_;
    for (int d = 0; d < D_; ++d) {
        float s = c[d];
        acc += (s / (1.f + __expf(-s))) * w[d];
    }
    out[b * (2 * D_) + o] = acc;
}

// LayerNorm over last dim 512; one wave32 per row; outputs f16 for WMMA A operand.
__global__ void ln_kernel(const float* __restrict__ x, const float* __restrict__ g,
                          const float* __restrict__ b, _Float16* __restrict__ out, int rows) {
    int gid  = blockIdx.x * blockDim.x + threadIdx.x;
    int row  = gid >> 5;
    int lane = threadIdx.x & 31;
    if (row >= rows) return;
    const float* xr = x + (size_t)row * D_;
    float v[16], s = 0.f, s2 = 0.f;
#pragma unroll
    for (int i = 0; i < 16; ++i) {
        float t = xr[lane + i * 32];
        v[i] = t; s += t; s2 += t * t;
    }
#pragma unroll
    for (int off = 16; off > 0; off >>= 1) {
        s  += __shfl_xor(s,  off, 32);
        s2 += __shfl_xor(s2, off, 32);
    }
    float mean = s * (1.f / D_);
    float var  = s2 * (1.f / D_) - mean * mean;
    float rstd = rsqrtf(var + 1e-5f);
    _Float16* orow = out + (size_t)row * D_;
#pragma unroll
    for (int i = 0; i < 16; ++i) {
        int c = lane + i * 32;
        orow[c] = (_Float16)((v[i] - mean) * rstd * g[c] + b[c]);
    }
}

// Softmax over rows of length L (32/128/256); in-place on f32 (final output =
// probabilities, as the reference returns post-softmax weights) + f16 copy.
__global__ void softmax_kernel(float* __restrict__ sc, _Float16* __restrict__ probs,
                               long long rows, int L) {
    long long gid = (long long)blockIdx.x * blockDim.x + threadIdx.x;
    long long row = gid >> 5;
    int lane = threadIdx.x & 31;
    if (row >= rows) return;
    float* r = sc + row * (long long)L;
    _Float16* ph = probs + row * (long long)L;
    int nIt = L >> 5;                 // 1, 4 or 8
    float v[8], mx = -3.4e38f;
    for (int i = 0; i < nIt; ++i) { v[i] = r[lane + i * 32]; mx = fmaxf(mx, v[i]); }
#pragma unroll
    for (int off = 16; off > 0; off >>= 1) mx = fmaxf(mx, __shfl_xor(mx, off, 32));
    float s = 0.f;
    for (int i = 0; i < nIt; ++i) { v[i] = __expf(v[i] - mx); s += v[i]; }
#pragma unroll
    for (int off = 16; off > 0; off >>= 1) s += __shfl_xor(s, off, 32);
    float inv = 1.f / s;
    for (int i = 0; i < nIt; ++i) {
        float p = v[i] * inv;
        r[lane + i * 32] = p;
        ph[lane + i * 32] = (_Float16)p;
    }
}

// ------------------------- GEMM parameter block ----------------------------
// epi: 0 -> Cf = alpha*acc (attention logits)
//      1 -> Ch = (f16)(acc + bias)   (Q/K/V, context)
//      2 -> Cf += (acc+bias)*(film_s+1)+film_sh  (out-proj FiLM residual)
//      3 -> Ch = (f16) gelu(acc+bias)            (FFN hidden)
struct GemmP {
    const _Float16* A; const _Float16* Bm;
    float* Cf; _Float16* Ch;
    const float* bias; const float* film;
    float alpha;
    int M, N, K, lda, ldb, ldc;
    int zd1, zd2;
    long long aO0, aO1, aO2, bO0, bO1, bO2, cO0, cO1, cO2;
    int epi;
};

__device__ __forceinline__ void gemm_epilogue(const GemmP& p, const v8f& acc,
                                              int mBase, int nBase, int hi, int lane,
                                              long long cB) {
#pragma unroll
    for (int r = 0; r < 8; ++r) {
        int m = mBase + r + 8 * hi;
        int n = nBase + (lane & 15);
        float v = acc[r];
        long long ci = cB + (long long)m * p.ldc + n;
        if (p.epi == 0) {
            p.Cf[ci] = v * p.alpha;
        } else if (p.epi == 1) {
            float bb = p.bias ? p.bias[n] : 0.f;
            p.Ch[ci] = (_Float16)(v + bb);
        } else if (p.epi == 2) {
            float bb = p.bias ? p.bias[n] : 0.f;
            float t  = v + bb;
            int  bidx = m / TOKPERB;
            float s  = p.film[bidx * (2 * D_) + n];
            float sh = p.film[bidx * (2 * D_) + D_ + n];
            p.Cf[ci] += t * (s + 1.f) + sh;
        } else {
            float bb = p.bias ? p.bias[n] : 0.f;
            float t  = v + bb;
            p.Ch[ci] = (_Float16)(0.5f * t * (1.f + erff(t * 0.70710678f)));
        }
    }
}

// ----------------- direct batched WMMA GEMM (attention) --------------------
// MODEB==0 (NT): B stored [N, K];  MODEB==1 (NN): B stored [K, N].
template <int MODEB>
__global__ __launch_bounds__(128) void gemm_wmma(GemmP p) {
    int lane  = threadIdx.x & 31;
    int wv    = threadIdx.x >> 5;
    int mTile = blockIdx.x * 4 + wv;
    if (mTile * 16 >= p.M) return;            // wave-uniform: EXEC stays all-ones
    int nTile = blockIdx.y;
    int z  = blockIdx.z;
    int z2 = z % p.zd2;
    int zr = z / p.zd2;
    int z1 = zr % p.zd1;
    int z0 = zr / p.zd1;
    long long aB = (long long)z0 * p.aO0 + (long long)z1 * p.aO1 + (long long)z2 * p.aO2;
    long long bB = (long long)z0 * p.bO0 + (long long)z1 * p.bO1 + (long long)z2 * p.bO2;
    long long cB = (long long)z0 * p.cO0 + (long long)z1 * p.cO1 + (long long)z2 * p.cO2;

    int mBase = mTile * 16, nBase = nTile * 16;
    int hi    = lane >> 4;
    int am    = mBase + (lane & 15);
    int aK8   = hi * 16;                      // A: K split {0..15} / {16..31}
    int bn    = nBase + (lane & 15);
    int bK16  = hi * 16;                      // B: K split {0..15} / {16..31}

    v8f acc = {};
    for (int kk = 0; kk < p.K; kk += 32) {
        const _Float16* ap = p.A + aB + (long long)am * p.lda + kk + aK8;
        h8 a0 = *(const h8*)ap;
        h8 a1 = *(const h8*)(ap + 8);
        v16h af = __builtin_shufflevector(a0, a1, 0,1,2,3,4,5,6,7,8,9,10,11,12,13,14,15);
        v16h bf;
        if (MODEB == 0) {
            const _Float16* bp = p.Bm + bB + (long long)bn * p.ldb + kk + bK16;
            h8 b0 = *(const h8*)bp;
            h8 b1 = *(const h8*)(bp + 8);
            bf = __builtin_shufflevector(b0, b1, 0,1,2,3,4,5,6,7,8,9,10,11,12,13,14,15);
        } else {
            const _Float16* bp = p.Bm + bB + (long long)(kk + bK16) * p.ldb + bn;
#pragma unroll
            for (int i = 0; i < 16; ++i) bf[i] = bp[(long long)i * p.ldb];
        }
        acc = __builtin_amdgcn_wmma_f32_16x16x32_f16(
            false, af, false, bf, (short)0, acc, false, false);
    }
    gemm_epilogue(p, acc, mBase, nBase, hi, lane, cB);
}

// ----------------- LDS-tiled WMMA GEMM (projections / FFN) -----------------
// NT only, batch=1. Block: 256 threads = 8 waves -> 64x64 C tile, BK=32.
// Wave w computes a 16(M) x 32(N) strip: 2 accumulators, A fragment reused
// across both WMMAs (back-to-back matrix-pipe issue). Double-buffered LDS,
// staged with async global->LDS copies.
#define BM 64
#define BN 64
#define BK 32

__global__ __launch_bounds__(256) void gemm_tiled(GemmP p) {
    __shared__ _Float16 sA[2][BM * BK];   // 4 KB per buffer
    __shared__ _Float16 sB[2][BN * BK];   // 4 KB per buffer
    int tid  = threadIdx.x;
    int lane = tid & 31, wv = tid >> 5;
    int mBlock = blockIdx.x * BM, nBlock = blockIdx.y * BN;
    int waveM = (wv >> 1) * 16;           // 4 M-strips
    int waveN = (wv & 1) * 32;            // 2 N-halves

    // staging assignment: 16B (8 halfs) per thread for each of A and B
    int arow = tid >> 2, ac8 = (tid & 3) * 8;
    const _Float16* gA = p.A  + (long long)(mBlock + arow) * p.lda + ac8;
    const _Float16* gB = p.Bm + (long long)(nBlock + arow) * p.ldb + ac8;

    auto stage = [&](int buf, int kk) {
#if USE_ASYNC
        __builtin_amdgcn_global_load_async_to_lds_b128(
            (AS_GLOBAL ai4*)(uintptr_t)(gA + kk),
            (AS_LOCAL  ai4*)(uint32_t)(uintptr_t)&sA[buf][arow * BK + ac8], 0, 0);
        __builtin_amdgcn_global_load_async_to_lds_b128(
            (AS_GLOBAL ai4*)(uintptr_t)(gB + kk),
            (AS_LOCAL  ai4*)(uint32_t)(uintptr_t)&sB[buf][arow * BK + ac8], 0, 0);
#else
        *(h8*)&sA[buf][arow * BK + ac8] = *(const h8*)(gA + kk);
        *(h8*)&sB[buf][arow * BK + ac8] = *(const h8*)(gB + kk);
#endif
        if (kk + 2 * BK < p.K) {             // prefetch 2 K-steps ahead, keep near
            __builtin_prefetch(gA + kk + 2 * BK, 0, 3);
            __builtin_prefetch(gB + kk + 2 * BK, 0, 3);
        }
    };

    int hi   = lane >> 4;
    int aK8  = hi * 16;                   // A: K split {0..15} / {16..31}
    int bK16 = hi * 16;                   // B: K split {0..15} / {16..31}
    int amL  = waveM + (lane & 15);
    int bnL0 = waveN + (lane & 15);
    int bnL1 = bnL0 + 16;

    int nSteps = p.K / BK;
    stage(0, 0);
    v8f acc0 = {}, acc1 = {};
    int cur = 0;
    for (int s = 0; s < nSteps; ++s) {
#if USE_ASYNC
        __builtin_amdgcn_s_wait_asynccnt(0);
#endif
        __syncthreads();
        if (s + 1 < nSteps) stage(cur ^ 1, (s + 1) * BK);
        const _Float16* pa = &sA[cur][amL * BK + aK8];
        h8 a0 = *(const h8*)pa;
        h8 a1 = *(const h8*)(pa + 8);
        v16h af = __builtin_shufflevector(a0, a1, 0,1,2,3,4,5,6,7,8,9,10,11,12,13,14,15);
        const _Float16* pb0 = &sB[cur][bnL0 * BK + bK16];
        h8 b00 = *(const h8*)pb0;
        h8 b01 = *(const h8*)(pb0 + 8);
        v16h bf0 = __builtin_shufflevector(b00, b01, 0,1,2,3,4,5,6,7,8,9,10,11,12,13,14,15);
        const _Float16* pb1 = &sB[cur][bnL1 * BK + bK16];
        h8 b10 = *(const h8*)pb1;
        h8 b11 = *(const h8*)(pb1 + 8);
        v16h bf1 = __builtin_shufflevector(b10, b11, 0,1,2,3,4,5,6,7,8,9,10,11,12,13,14,15);
        acc0 = __builtin_amdgcn_wmma_f32_16x16x32_f16(
            false, af, false, bf0, (short)0, acc0, false, false);
        acc1 = __builtin_amdgcn_wmma_f32_16x16x32_f16(
            false, af, false, bf1, (short)0, acc1, false, false);
        cur ^= 1;
    }
    gemm_epilogue(p, acc0, mBlock + waveM, nBlock + waveN,      hi, lane, 0);
    gemm_epilogue(p, acc1, mBlock + waveM, nBlock + waveN + 16, hi, lane, 0);
}

static inline void gemmLaunch(const GemmP& p, int zTot, int modeB, hipStream_t s) {
    dim3 g((p.M + 63) / 64, p.N / 16, zTot), b(128);
    if (modeB == 0) gemm_wmma<0><<<g, b, 0, s>>>(p);
    else            gemm_wmma<1><<<g, b, 0, s>>>(p);
}
static inline void gemmTiledLaunch(const GemmP& p, hipStream_t s) {
    dim3 g(p.M / BM, p.N / BN), b(256);
    gemm_tiled<<<g, b, 0, s>>>(p);
}

// ------------------------------ orchestration ------------------------------

extern "C" void kernel_launch(void* const* d_in, const int* in_sizes, int n_in,
                              void* d_out, int out_size, void* d_ws, size_t ws_size,
                              hipStream_t stream) {
    const float* x    = (const float*)d_in[0];
    const float* memo = (const float*)d_in[1];
    const float* cond = (const float*)d_in[2];
    const float* W_ta[4] = {(const float*)d_in[3], (const float*)d_in[4], (const float*)d_in[5], (const float*)d_in[6]};
    const float* b_ta[4] = {(const float*)d_in[7], (const float*)d_in[8], (const float*)d_in[9], (const float*)d_in[10]};
    const float* W_sa[4] = {(const float*)d_in[11], (const float*)d_in[12], (const float*)d_in[13], (const float*)d_in[14]};
    const float* b_sa[4] = {(const float*)d_in[15], (const float*)d_in[16], (const float*)d_in[17], (const float*)d_in[18]};
    const float* W_ca[4] = {(const float*)d_in[19], (const float*)d_in[20], (const float*)d_in[21], (const float*)d_in[22]};
    const float* b_ca[4] = {(const float*)d_in[23], (const float*)d_in[24], (const float*)d_in[25], (const float*)d_in[26]};
    const float* temp_ng = (const float*)d_in[27]; const float* temp_nb = (const float*)d_in[28];
    const float* skel_ng = (const float*)d_in[29]; const float* skel_nb = (const float*)d_in[30];
    const float* cs_ng   = (const float*)d_in[31]; const float* cs_nb   = (const float*)d_in[32];
    const float* ct_ng   = (const float*)d_in[33]; const float* ct_nb   = (const float*)d_in[34];
    const float* ffn_ng  = (const float*)d_in[35]; const float* ffn_nb  = (const float*)d_in[36];
    const float* ff_W1 = (const float*)d_in[37]; const float* ff_b1 = (const float*)d_in[38];
    const float* ff_W2 = (const float*)d_in[39]; const float* ff_b2 = (const float*)d_in[40];
    const float* filmW[4] = {(const float*)d_in[41], (const float*)d_in[43], (const float*)d_in[45], (const float*)d_in[47]};
    const float* filmB[4] = {(const float*)d_in[42], (const float*)d_in[44], (const float*)d_in[46], (const float*)d_in[48]};

    // output regions: x, sa_w, ta_w, ca_w (flat concat)
    float* xo  = (float*)d_out;
    float* saw = xo + 16777216LL;   // [B*T, H, J, J]
    float* taw = xo + 25165824LL;   // [B*J, H, T, T]
    float* caw = xo + 58720256LL;   // [B, H, T*J, M]

    // workspace layout
    char* w = (char*)d_ws;
    _Float16* wh = (_Float16*)w;                       // 12 x 262144 attn mats
    _Float16* whA[12];
    for (int i = 0; i < 12; ++i) whA[i] = wh + (long long)i * 262144;
    _Float16* ffW1h = (_Float16*)(w + 6291456);        // [2048,512]
    _Float16* ffW2h = (_Float16*)(w + 8388608);        // [512,2048]
    float*    filmO = (float*)(w + 10485760);          // 4 x [8,1024]
    _Float16* n_h   = (_Float16*)(w + 10616832);       // [32768,512]
    _Float16* mem_h = (_Float16*)(w + 44171264);       // [2048,512]
    _Float16* q_h   = (_Float16*)(w + 46268416);       // [32768,512]
    _Float16* k_h   = (_Float16*)(w + 79822848);
    _Float16* v_h   = (_Float16*)(w + 113377280);
    _Float16* ctx_h = (_Float16*)(w + 146931712);
    _Float16* big   = (_Float16*)(w + 180486144);      // shared probs / FFN hidden (134MB)

    // residual state lives in d_out x-region
    (void)hipMemcpyAsync(xo, x, (size_t)NTOK * D_ * sizeof(float), hipMemcpyDeviceToDevice, stream);

    auto cvt = [&](const float* src, _Float16* dst, long long n) {
        cvt_f32_f16<<<dim3((unsigned)((n + 255) / 256)), 256, 0, stream>>>(src, dst, n);
    };
    for (int i = 0; i < 4; ++i) cvt(W_ta[i], whA[i],     262144);
    for (int i = 0; i < 4; ++i) cvt(W_sa[i], whA[4 + i], 262144);
    for (int i = 0; i < 4; ++i) cvt(W_ca[i], whA[8 + i], 262144);
    cvt(ff_W1, ffW1h, 1048576);
    cvt(ff_W2, ffW2h, 1048576);

    for (int f = 0; f < 4; ++f)
        film_kernel<<<dim3(4, 8), 256, 0, stream>>>(cond, filmW[f], filmB[f], filmO + f * 8192);

    auto ln = [&](const float* in, const float* gg, const float* bb, _Float16* out, int rows) {
        ln_kernel<<<dim3((rows + 7) / 8), 256, 0, stream>>>(in, gg, bb, out, rows);
    };
    auto proj = [&](const _Float16* A, int Mrows, const _Float16* Wm, const float* bias, _Float16* outH) {
        GemmP p{}; p.A = A; p.Bm = Wm; p.Ch = outH; p.bias = bias; p.alpha = 1.f;
        p.M = Mrows; p.N = D_; p.K = D_; p.lda = D_; p.ldb = D_; p.ldc = D_;
        p.zd1 = 1; p.zd2 = 1; p.epi = 1;
        gemmTiledLaunch(p, stream);
    };
    auto outproj = [&](const _Float16* Wm, const float* bias, const float* film) {
        GemmP p{}; p.A = ctx_h; p.Bm = Wm; p.Cf = xo; p.bias = bias; p.film = film; p.alpha = 1.f;
        p.M = NTOK; p.N = D_; p.K = D_; p.lda = D_; p.ldb = D_; p.ldc = D_;
        p.zd1 = 1; p.zd2 = 1; p.epi = 2;
        gemmTiledLaunch(p, stream);
    };

    // ================= Phase 1: temporal attention (over T, batch B*J) =====
    ln(xo, temp_ng, temp_nb, n_h, NTOK);
    proj(n_h, NTOK, whA[0], b_ta[0], q_h);
    proj(n_h, NTOK, whA[1], b_ta[1], k_h);
    proj(n_h, NTOK, whA[2], b_ta[2], v_h);
    {   // logits: z=(b,j,h); Q/K rows strided by J*D
        GemmP p{}; p.A = q_h; p.Bm = k_h; p.Cf = taw; p.alpha = 0.125f;
        p.M = T_; p.N = T_; p.K = 64; p.lda = J_ * D_; p.ldb = J_ * D_; p.ldc = T_;
        p.zd1 = J_; p.zd2 = H_;
        p.aO0 = 2097152; p.aO1 = D_; p.aO2 = 64;
        p.bO0 = 2097152; p.bO1 = D_; p.bO2 = 64;
        p.cO0 = 4194304; p.cO1 = 131072; p.cO2 = 16384;
        p.epi = 0;
        gemmLaunch(p, B_ * J_ * H_, 0, stream);
    }
    softmax_kernel<<<dim3(32768), 256, 0, stream>>>(taw, big, (long long)B_ * J_ * H_ * T_, T_);
    {   // context: P[128,128] @ V[128,64] (NN)
        GemmP p{}; p.A = big; p.Bm = v_h; p.Ch = ctx_h; p.alpha = 1.f;
        p.M = T_; p.N = 64; p.K = T_; p.lda = T_; p.ldb = J_ * D_; p.ldc = J_ * D_;
        p.zd1 = J_; p.zd2 = H_;
        p.aO0 = 4194304; p.aO1 = 131072; p.aO2 = 16384;
        p.bO0 = 2097152; p.bO1 = D_; p.bO2 = 64;
        p.cO0 = 2097152; p.cO1 = D_; p.cO2 = 64;
        p.epi = 1;
        gemmLaunch(p, B_ * J_ * H_, 1, stream);
    }
    outproj(whA[3], b_ta[3], filmO + 0 * 8192);

    // ================= Phase 2: skeletal attention (over J, batch B*T) =====
    ln(xo, skel_ng, skel_nb, n_h, NTOK);
    proj(n_h, NTOK, whA[4], b_sa[0], q_h);
    proj(n_h, NTOK, whA[5], b_sa[1], k_h);
    proj(n_h, NTOK, whA[6], b_sa[2], v_h);
    {   // logits: z=(b,t,h); rows strided by D
        GemmP p{}; p.A = q_h; p.Bm = k_h; p.Cf = saw; p.alpha = 0.125f;
        p.M = J_; p.N = J_; p.K = 64; p.lda = D_; p.ldb = D_; p.ldc = J_;
        p.zd1 = T_; p.zd2 = H_;
        p.aO0 = 2097152; p.aO1 = 16384; p.aO2 = 64;
        p.bO0 = 2097152; p.bO1 = 16384; p.bO2 = 64;
        p.cO0 = 1048576; p.cO1 = 8192; p.cO2 = 1024;
        p.epi = 0;
        gemmLaunch(p, B_ * T_ * H_, 0, stream);
    }
    softmax_kernel<<<dim3(32768), 256, 0, stream>>>(saw, big, (long long)B_ * T_ * H_ * J_, J_);
    {   // context: P[32,32] @ V[32,64] (NN)
        GemmP p{}; p.A = big; p.Bm = v_h; p.Ch = ctx_h; p.alpha = 1.f;
        p.M = J_; p.N = 64; p.K = J_; p.lda = J_; p.ldb = D_; p.ldc = D_;
        p.zd1 = T_; p.zd2 = H_;
        p.aO0 = 1048576; p.aO1 = 8192; p.aO2 = 1024;
        p.bO0 = 2097152; p.bO1 = 16384; p.bO2 = 64;
        p.cO0 = 2097152; p.cO1 = 16384; p.cO2 = 64;
        p.epi = 1;
        gemmLaunch(p, B_ * T_ * H_, 1, stream);
    }
    outproj(whA[7], b_sa[3], filmO + 1 * 8192);

    // ================= Phase 3: cross attention to memory ==================
    ln(xo, cs_ng, cs_nb, n_h, NTOK);
    ln(memo, ct_ng, ct_nb, mem_h, B_ * M_);
    proj(n_h, NTOK, whA[8], b_ca[0], q_h);
    proj(mem_h, B_ * M_, whA[9],  b_ca[1], k_h);
    proj(mem_h, B_ * M_, whA[10], b_ca[2], v_h);
    {   // logits: z=(b,h); Q [4096,64], K [256,64]
        GemmP p{}; p.A = q_h; p.Bm = k_h; p.Cf = caw; p.alpha = 0.125f;
        p.M = TOKPERB; p.N = M_; p.K = 64; p.lda = D_; p.ldb = D_; p.ldc = M_;
        p.zd1 = H_; p.zd2 = 1;
        p.aO0 = 2097152; p.aO1 = 64; p.aO2 = 0;
        p.bO0 = 131072;  p.bO1 = 64; p.bO2 = 0;
        p.cO0 = 8388608; p.cO1 = 1048576; p.cO2 = 0;
        p.epi = 0;
        gemmLaunch(p, B_ * H_, 0, stream);
    }
    softmax_kernel<<<dim3(32768), 256, 0, stream>>>(caw, big, (long long)B_ * H_ * TOKPERB, M_);
    {   // context: P[4096,256] @ V[256,64] (NN)
        GemmP p{}; p.A = big; p.Bm = v_h; p.Ch = ctx_h; p.alpha = 1.f;
        p.M = TOKPERB; p.N = 64; p.K = M_; p.lda = M_; p.ldb = D_; p.ldc = D_;
        p.zd1 = H_; p.zd2 = 1;
        p.aO0 = 8388608; p.aO1 = 1048576; p.aO2 = 0;
        p.bO0 = 131072;  p.bO1 = 64; p.bO2 = 0;
        p.cO0 = 2097152; p.cO1 = 64; p.cO2 = 0;
        p.epi = 1;
        gemmLaunch(p, B_ * H_, 1, stream);
    }
    outproj(whA[11], b_ca[3], filmO + 2 * 8192);

    // ================= Phase 4: FFN ========================================
    ln(xo, ffn_ng, ffn_nb, n_h, NTOK);
    {   // hidden = gelu(n @ W1.T + b1) -> f16 in `big`
        GemmP p{}; p.A = n_h; p.Bm = ffW1h; p.Ch = big; p.bias = ff_b1; p.alpha = 1.f;
        p.M = NTOK; p.N = FF_; p.K = D_; p.lda = D_; p.ldb = D_; p.ldc = FF_;
        p.zd1 = 1; p.zd2 = 1; p.epi = 3;
        gemmTiledLaunch(p, stream);
    }
    {   // x += (hidden @ W2.T + b2) * (s+1) + sh
        GemmP p{}; p.A = big; p.Bm = ffW2h; p.Cf = xo; p.bias = ff_b2;
        p.film = filmO + 3 * 8192; p.alpha = 1.f;
        p.M = NTOK; p.N = D_; p.K = FF_; p.lda = FF_; p.ldb = FF_; p.ldc = D_;
        p.zd1 = 1; p.zd2 = 1; p.epi = 2;
        gemmTiledLaunch(p, stream);
    }
}
